// GAT_17308718202891
// MI455X (gfx1250) — matrix operand order repaired and checked by reading, not compiled, not run on hardware
//
#include <hip/hip_runtime.h>
#include <hip/hip_bf16.h>
#include <math.h>

#define NNODES 20000
#define NEDGES 320000
#define DIN    256
#define DHID   512   // H * DH == H * DOUT
#define HEADS  8
#define CH     64    // per-head channels
#define NEG_SLOPE 0.2f

typedef __attribute__((ext_vector_type(16))) _Float16 v16h;
typedef __attribute__((ext_vector_type(8)))  _Float16 v8h;
typedef __attribute__((ext_vector_type(4)))  _Float16 v4h;
typedef __attribute__((ext_vector_type(8)))  float    v8f;

// ---------------- utility kernels ----------------
__global__ __launch_bounds__(256) void f32_to_f16_kernel(const float* __restrict__ in,
                                                         _Float16* __restrict__ out, int n) {
  int i = blockIdx.x * 256 + threadIdx.x;
  if (i < n) out[i] = (_Float16)in[i];
}

// weights: W[k][n] (row-major KxN) -> Wt[n][k] (row-major NxK), f32 -> f16
__global__ __launch_bounds__(256) void f32_to_f16_transpose_kernel(
    const float* __restrict__ in, _Float16* __restrict__ out, int K, int N) {
  int i = blockIdx.x * 256 + threadIdx.x;
  if (i >= K * N) return;
  int k = i / N, n = i % N;
  out[(size_t)n * K + k] = (_Float16)in[i];
}

__global__ __launch_bounds__(256) void fill_kernel(float* __restrict__ p, float v, int n) {
  int i = blockIdx.x * 256 + threadIdx.x;
  if (i < n) p[i] = v;
}

// ---------------- WMMA GEMM:  C[M,512] = A[M,K]_f16 @ Bt[512,K]_f16^T + bias ----------------
// 256 threads = 8 waves; block tile 32x64; wave (w>>2, w&3) owns a 16x16 D tile; K-step 32.
// A staged in LDS (reused by 4 waves / row group); B read directly from transposed
// weights in global (L2-resident), two b128 loads per lane per step.
template <int K>
__global__ __launch_bounds__(256) void gemm_wmma_kernel(
    const _Float16* __restrict__ A, const _Float16* __restrict__ Bt,
    const float* __restrict__ bias, float* __restrict__ C, int M)
{
  constexpr int N = DHID;
  __shared__ __align__(16) _Float16 As[32][32];

  const int tid  = threadIdx.x;
  const int wave = tid >> 5;
  const int lane = tid & 31;
  const int block_row = blockIdx.x * 32;
  const int block_col = blockIdx.y * 64;
  const int m0 = (wave >> 2) * 16;   // 0 or 16
  const int n0 = (wave & 3) * 16;    // 0,16,32,48

  // staging coords: each thread moves 4 contiguous halfs (8B) of the 32x32 A tile
  const int sr = tid >> 3;            // row 0..31
  const int sc = (tid & 7) * 4;       // col 0..28
  const _Float16* agp = A + (size_t)(block_row + sr) * K + sc;

  // fragment coords (ISA 16-bit A/B layouts)
  const int fm  = m0 + (lane & 15);
  const int fkb = (lane >> 4) * 8;          // A: K sub-block 0 or 8
  const _Float16* bgp = Bt + (size_t)(block_col + n0 + (lane & 15)) * K
                           + ((lane >> 4) << 3);   // B: K sub-block 0 or 8

  v8f acc = {};
  #pragma unroll
  for (int k0 = 0; k0 < K; k0 += 32) {
    // stage A tile (vector 8B copy, no guards: M is a multiple of 32)
    *reinterpret_cast<v4h*>(&As[sr][sc]) =
        *reinterpret_cast<const v4h*>(agp + k0);
    if (k0 + 32 < K)
      __builtin_prefetch(agp + k0 + 32, 0, 3);
    __syncthreads();

    // A fragment: two contiguous 16B runs per lane -> ds_load_b128 x2
    v8h a_lo = *reinterpret_cast<const v8h*>(&As[fm][fkb]);
    v8h a_hi = *reinterpret_cast<const v8h*>(&As[fm][16 + fkb]);
    v16h af = __builtin_shufflevector(a_lo, a_hi,
        0,1,2,3,4,5,6,7,8,9,10,11,12,13,14,15);

    // B fragment: two contiguous 16B runs per lane from transposed weights
    const v8h* bp = reinterpret_cast<const v8h*>(bgp + k0);
    v8h b_lo = bp[0];
    v8h b_hi = bp[2];
    v16h bfv = __builtin_shufflevector(b_lo, b_hi,
        0,1,2,3,4,5,6,7,8,9,10,11,12,13,14,15);

    acc = __builtin_amdgcn_wmma_f32_16x16x32_f16(false, af, false, bfv,
                                                 (short)0, acc, false, false);
    __syncthreads();
  }

  // D layout: VGPR j -> row m0 + j + 8*(lane>=16), col n0 + lane%16
  const int col = block_col + n0 + (lane & 15);
  const float bv = bias[col];
  const int row_base = block_row + m0 + ((lane >> 4) << 3);
  #pragma unroll
  for (int j = 0; j < 8; ++j)
    C[(size_t)(row_base + j) * N + col] = acc[j] + bv;
}

// ---------------- edge kernels (one thread per (edge, head)) ----------------
__device__ __forceinline__ void atomic_max_f32(float* addr, float v) {
  if (v >= 0.f) atomicMax((int*)addr, __float_as_int(v));
  else          atomicMin((unsigned int*)addr, __float_as_uint(v));
}

__global__ __launch_bounds__(256) void edge_logits_kernel(
    const int* __restrict__ ei, const float* __restrict__ l, const float* __restrict__ r,
    const float* __restrict__ att, float* __restrict__ logits,
    float* __restrict__ nmax, int E, int N)
{
  int t = blockIdx.x * 256 + threadIdx.x;
  int total = (E + N) * HEADS;
  if (t >= total) return;
  int e = t >> 3, h = t & 7;
  int src = (e < E) ? ei[e]     : (e - E);
  int dst = (e < E) ? ei[E + e] : (e - E);
  const float4* lp = (const float4*)(l + ((long)src * HEADS + h) * CH);
  const float4* rp = (const float4*)(r + ((long)dst * HEADS + h) * CH);
  const float4* ap = (const float4*)(att + h * CH);
  float s = 0.f;
  #pragma unroll
  for (int i = 0; i < CH / 4; ++i) {
    float4 lv = lp[i], rv = rp[i], av = ap[i];
    float v;
    v = lv.x + rv.x; v = v > 0.f ? v : NEG_SLOPE * v; s += av.x * v;
    v = lv.y + rv.y; v = v > 0.f ? v : NEG_SLOPE * v; s += av.y * v;
    v = lv.z + rv.z; v = v > 0.f ? v : NEG_SLOPE * v; s += av.z * v;
    v = lv.w + rv.w; v = v > 0.f ? v : NEG_SLOPE * v; s += av.w * v;
  }
  logits[t] = s;
  atomic_max_f32(&nmax[dst * HEADS + h], s);
}

__global__ __launch_bounds__(256) void edge_exp_kernel(
    const int* __restrict__ ei, float* __restrict__ logits,
    const float* __restrict__ nmax, float* __restrict__ denom, int E, int N)
{
  int t = blockIdx.x * 256 + threadIdx.x;
  int total = (E + N) * HEADS;
  if (t >= total) return;
  int e = t >> 3, h = t & 7;
  int dst = (e < E) ? ei[E + e] : (e - E);
  float ex = __expf(logits[t] - nmax[dst * HEADS + h]);
  logits[t] = ex;
  atomicAdd(&denom[dst * HEADS + h], ex);
}

__global__ __launch_bounds__(256) void edge_scatter_kernel(
    const int* __restrict__ ei, const float* __restrict__ l,
    const float* __restrict__ logits, const float* __restrict__ denom,
    float* __restrict__ acc, int E, int N)
{
  int t = blockIdx.x * 256 + threadIdx.x;
  int total = (E + N) * HEADS;
  if (t >= total) return;
  int e = t >> 3, h = t & 7;
  int src = (e < E) ? ei[e]     : (e - E);
  int dst = (e < E) ? ei[E + e] : (e - E);
  float alpha = logits[t] / denom[dst * HEADS + h];
  const float4* lp = (const float4*)(l + ((long)src * HEADS + h) * CH);
  float* op = acc + ((long)dst * HEADS + h) * CH;
  #pragma unroll
  for (int i = 0; i < CH / 4; ++i) {
    float4 lv = lp[i];
    atomicAdd(&op[4 * i + 0], lv.x * alpha);
    atomicAdd(&op[4 * i + 1], lv.y * alpha);
    atomicAdd(&op[4 * i + 2], lv.z * alpha);
    atomicAdd(&op[4 * i + 3], lv.w * alpha);
  }
}

// layer-1 epilogue: h = elu(acc + bias); emit f16 for next layer's WMMA GEMM
__global__ __launch_bounds__(256) void bias_elu_f16_kernel(
    const float* __restrict__ acc, const float* __restrict__ bias,
    _Float16* __restrict__ out16, int n, int D)
{
  int i = blockIdx.x * 256 + threadIdx.x;
  if (i >= n) return;
  float v = acc[i] + bias[i & (D - 1)];
  v = v > 0.f ? v : (__expf(v) - 1.0f);
  out16[i] = (_Float16)v;
}

// final: h2 = acc + bias2 ; out[0:N*D) = h2 ; out[N*D:2N*D) = log_softmax(h2)
__global__ __launch_bounds__(256) void out_logsoftmax_kernel(
    const float* __restrict__ acc, const float* __restrict__ bias,
    float* __restrict__ out, int N, int D)
{
  __shared__ float red[256];
  int row = blockIdx.x, tid = threadIdx.x;
  const float* a = acc + (long)row * D;
  float lm = -INFINITY;
  for (int c = tid; c < D; c += 256) lm = fmaxf(lm, a[c] + bias[c]);
  red[tid] = lm; __syncthreads();
  for (int s = 128; s > 0; s >>= 1) { if (tid < s) red[tid] = fmaxf(red[tid], red[tid + s]); __syncthreads(); }
  float m = red[0]; __syncthreads();
  float ls = 0.f;
  for (int c = tid; c < D; c += 256) ls += __expf(a[c] + bias[c] - m);
  red[tid] = ls; __syncthreads();
  for (int s = 128; s > 0; s >>= 1) { if (tid < s) red[tid] += red[tid + s]; __syncthreads(); }
  float lse = m + __logf(red[0]);
  float* oh = out + (long)row * D;
  float* ol = out + (long)N * D + (long)row * D;
  for (int c = tid; c < D; c += 256) { float v = a[c] + bias[c]; oh[c] = v; ol[c] = v - lse; }
}

// ---------------- host ----------------
extern "C" void kernel_launch(void* const* d_in, const int* in_sizes, int n_in,
                              void* d_out, int out_size, void* d_ws, size_t ws_size,
                              hipStream_t stream) {
  const float* x     = (const float*)d_in[0];
  const int*   ei    = (const int*)  d_in[1];
  const float* Wl1   = (const float*)d_in[2];
  const float* bl1   = (const float*)d_in[3];
  const float* Wr1   = (const float*)d_in[4];
  const float* br1   = (const float*)d_in[5];
  const float* att1  = (const float*)d_in[6];
  const float* bias1 = (const float*)d_in[7];
  const float* Wl2   = (const float*)d_in[8];
  const float* bl2   = (const float*)d_in[9];
  const float* Wr2   = (const float*)d_in[10];
  const float* br2   = (const float*)d_in[11];
  const float* att2  = (const float*)d_in[12];
  const float* bias2 = (const float*)d_in[13];
  float* out = (float*)d_out;
  (void)in_sizes; (void)n_in; (void)out_size; (void)ws_size;

  const int N = NNODES, E = NEDGES;
  const int Etot = E + N;
  const int totEH = Etot * HEADS;

  char* ws = (char*)d_ws;
  size_t off = 0;
  auto alloc = [&](size_t bytes) -> void* {
    void* p = ws + off; off = (off + bytes + 255) & ~(size_t)255; return p;
  };
  float*    buf_l   = (float*)   alloc((size_t)N * DHID * 4);
  float*    buf_r   = (float*)   alloc((size_t)N * DHID * 4);
  float*    buf_acc = (float*)   alloc((size_t)N * DHID * 4);
  float*    buf_lg  = (float*)   alloc((size_t)Etot * HEADS * 4);
  float*    buf_max = (float*)   alloc((size_t)N * HEADS * 4);
  float*    buf_den = (float*)   alloc((size_t)N * HEADS * 4);
  _Float16* x16     = (_Float16*)alloc((size_t)N * DIN * 2);
  _Float16* h16     = (_Float16*)alloc((size_t)N * DHID * 2);
  _Float16* w16t    = (_Float16*)alloc((size_t)DHID * DHID * 2);  // transposed weights

  dim3 blk(256);
  auto cdiv = [](int a, int b) { return (a + b - 1) / b; };
  dim3 ggrid(N / 32, DHID / 64);   // 625 x 8 tiles

  // stage x in f16
  f32_to_f16_kernel<<<cdiv(N * DIN, 256), blk, 0, stream>>>(x, x16, N * DIN);

  // ===== Layer 1: l = x@Wl1+bl1, r = x@Wr1+br1 =====
  f32_to_f16_transpose_kernel<<<cdiv(DIN * DHID, 256), blk, 0, stream>>>(Wl1, w16t, DIN, DHID);
  gemm_wmma_kernel<DIN><<<ggrid, blk, 0, stream>>>(x16, w16t, bl1, buf_l, N);
  f32_to_f16_transpose_kernel<<<cdiv(DIN * DHID, 256), blk, 0, stream>>>(Wr1, w16t, DIN, DHID);
  gemm_wmma_kernel<DIN><<<ggrid, blk, 0, stream>>>(x16, w16t, br1, buf_r, N);

  fill_kernel<<<cdiv(N * HEADS, 256), blk, 0, stream>>>(buf_max, -INFINITY, N * HEADS);
  fill_kernel<<<cdiv(N * HEADS, 256), blk, 0, stream>>>(buf_den, 0.f, N * HEADS);
  fill_kernel<<<cdiv(N * DHID, 256), blk, 0, stream>>>(buf_acc, 0.f, N * DHID);

  edge_logits_kernel <<<cdiv(totEH, 256), blk, 0, stream>>>(ei, buf_l, buf_r, att1, buf_lg, buf_max, E, N);
  edge_exp_kernel    <<<cdiv(totEH, 256), blk, 0, stream>>>(ei, buf_lg, buf_max, buf_den, E, N);
  edge_scatter_kernel<<<cdiv(totEH, 256), blk, 0, stream>>>(ei, buf_l, buf_lg, buf_den, buf_acc, E, N);

  bias_elu_f16_kernel<<<cdiv(N * DHID, 256), blk, 0, stream>>>(buf_acc, bias1, h16, N * DHID, DHID);

  // ===== Layer 2: l = h@Wl2+bl2, r = h@Wr2+br2 =====
  f32_to_f16_transpose_kernel<<<cdiv(DHID * DHID, 256), blk, 0, stream>>>(Wl2, w16t, DHID, DHID);
  gemm_wmma_kernel<DHID><<<ggrid, blk, 0, stream>>>(h16, w16t, bl2, buf_l, N);
  f32_to_f16_transpose_kernel<<<cdiv(DHID * DHID, 256), blk, 0, stream>>>(Wr2, w16t, DHID, DHID);
  gemm_wmma_kernel<DHID><<<ggrid, blk, 0, stream>>>(h16, w16t, br2, buf_r, N);

  fill_kernel<<<cdiv(N * HEADS, 256), blk, 0, stream>>>(buf_max, -INFINITY, N * HEADS);
  fill_kernel<<<cdiv(N * HEADS, 256), blk, 0, stream>>>(buf_den, 0.f, N * HEADS);
  fill_kernel<<<cdiv(N * DHID, 256), blk, 0, stream>>>(buf_acc, 0.f, N * DHID);

  edge_logits_kernel <<<cdiv(totEH, 256), blk, 0, stream>>>(ei, buf_l, buf_r, att2, buf_lg, buf_max, E, N);
  edge_exp_kernel    <<<cdiv(totEH, 256), blk, 0, stream>>>(ei, buf_lg, buf_max, buf_den, E, N);
  edge_scatter_kernel<<<cdiv(totEH, 256), blk, 0, stream>>>(ei, buf_l, buf_lg, buf_den, buf_acc, E, N);

  // ===== outputs: h2 and log_softmax(h2) =====
  out_logsoftmax_kernel<<<N, blk, 0, stream>>>(buf_acc, bias2, out, N, DHID);
}
